// CTM_18459769438494
// MI455X (gfx1250) — matrix operand order repaired and checked by reading, not compiled, not run on hardware
//
#include <hip/hip_runtime.h>
#include <hip/hip_bf16.h>
#include <hip/hip_fp16.h>

typedef __attribute__((ext_vector_type(16))) _Float16 v16h;
typedef __attribute__((ext_vector_type(16))) __bf16   v16bf;
typedef __attribute__((ext_vector_type(8)))  float    v8f;

#define B_ 4
#define H_ 56
#define W_ 56
#define N_ 3136
#define CI_ 64
#define CO_ 128
#define K_ 5
#define M_ 784
#define H2_ 28
#define W2_ 28
#define LN_EPS 1e-5f
#define INV_SQRT_C 0.08838834764831845f   // 1/sqrt(128)

// ---- workspace layout (float units) ----
#define OFF_CONV   0
#define SZ_CONV    (B_*H2_*W2_*CO_)              // 401408
#define OFF_XNB    (OFF_CONV + SZ_CONV)          // bf16 xn copy: B*N*CO bf16
#define SZ_XNB     (B_*N_*CO_/2)                 // 802816 float-equiv
#define OFF_SQ     (OFF_XNB + SZ_XNB)            // B*N row sq-norms
#define SZ_BN      (B_*N_)
#define OFF_WEIGHT (OFF_SQ + SZ_BN)
#define OFF_DENS   (OFF_WEIGHT + SZ_BN)
#define OFF_DMAX   (OFF_DENS + SZ_BN)            // B_ uints (+pad)
#define OFF_SCORE  (OFF_DMAX + 16)
#define OFF_IDOWN  (OFF_SCORE + SZ_BN)           // B*M ints
#define OFF_ALLW   (OFF_IDOWN + B_*M_)
#define OFF_DIST   (OFF_ALLW + B_*M_)            // B*N*N floats (157 MB, L2-resident)

// ---- d_out layout (floats): x_merged | xn | token_score | idx_cluster ----
#define OUT_XM  0
#define OUT_XN  (B_*M_*CO_)
#define OUT_TS  (OUT_XN + B_*N_*CO_)
#define OUT_IDX (OUT_TS + B_*N_)

// ---------------- init: zero accumulators ----------------
__global__ void CTM_init(float* __restrict__ xm, float* __restrict__ allw,
                         unsigned int* __restrict__ dmax) {
  int i = blockIdx.x * blockDim.x + threadIdx.x;
  if (i < B_*M_*CO_) xm[i] = 0.f;
  if (i < B_*M_)     allw[i] = 0.f;
  if (i < B_)        dmax[i] = 0u;
}

// ---------------- 3x3 stride-2 conv (NHWC,HWIO) ----------------
__global__ void CTM_conv(const float* __restrict__ x, const float* __restrict__ w,
                         const float* __restrict__ bias, float* __restrict__ cmap) {
  int idx = blockIdx.x * blockDim.x + threadIdx.x;
  if (idx >= B_*H2_*W2_*CO_) return;
  int oc = idx & (CO_-1);
  int t  = idx >> 7;
  int ow = t % W2_; t /= W2_;
  int oh = t % H2_; int b = t / H2_;
  float acc = bias[oc];
  for (int kh = 0; kh < 3; ++kh) {
    int ih = oh*2 - 1 + kh;
    if (ih < 0 || ih >= H_) continue;
    for (int kw = 0; kw < 3; ++kw) {
      int iw = ow*2 - 1 + kw;
      if (iw < 0 || iw >= W_) continue;
      const float* xp = x + ((size_t)b*N_ + ih*W_ + iw) * CI_;
      const float* wp = w + (size_t)((kh*3 + kw) * CI_) * CO_ + oc;
      #pragma unroll 8
      for (int ci = 0; ci < CI_; ++ci) acc += xp[ci] * wp[(size_t)ci * CO_];
    }
  }
  cmap[idx] = acc;
}

// ------- fused: skip GEMM (f16 WMMA) + map2token add + LN + score/exp + sq -------
__global__ void __launch_bounds__(256) CTM_fused_ln(
    const float* __restrict__ x, const float* __restrict__ skip_w,
    const float* __restrict__ cmap,
    const float* __restrict__ ln_g, const float* __restrict__ ln_b,
    const float* __restrict__ score_w, const float* __restrict__ score_b,
    float* __restrict__ xn_out, float* __restrict__ ts_out,
    __bf16* __restrict__ xnb, float* __restrict__ sq, float* __restrict__ weight) {
  __shared__ _Float16 Ash[16][CI_];
  __shared__ float xbuf[16][CO_];
  __shared__ float mu_s[16], rs_s[16];
  const int tid  = threadIdx.x;
  const int lane = tid & 31;
  const int wv   = tid >> 5;          // 8 waves: one 16x16 C_OUT tile each
  const int t0   = blockIdx.x * 16;   // 16 tokens per block (never crosses batch)

  // stage A: 16 tokens x 64ch as f16
  #pragma unroll
  for (int e = 0; e < 4; ++e) {
    int i = tid + 256 * e;
    int r = i >> 6, c = i & 63;
    Ash[r][c] = (_Float16)x[(size_t)(t0 + r) * CI_ + c];
  }
  __syncthreads();

  const int oc0 = wv * 16;
  v8f acc = {};
  #pragma unroll
  for (int s = 0; s < 2; ++s) {       // K = 64 = 2 x 32
    const int ks = s * 32;
    v16h a;                            // A frag: 16x32 f16, CDNA5 layout
    {
      int m  = lane & 15;
      int kb = (lane < 16) ? 0 : 8;
      #pragma unroll
      for (int e = 0; e < 16; ++e) {
        int r = e >> 1, h = e & 1;
        int kl = (r < 4) ? (kb + 2*r + h) : (16 + kb + 2*(r-4) + h);
        a[e] = Ash[m][ks + kl];
      }
    }
    v16h bf;                           // B frag: B[k][n] = skip_w[oc][k]
    {
      int n  = lane & 15;
      int kb = (lane < 16) ? 0 : 8;
      #pragma unroll
      for (int e = 0; e < 16; ++e)
        bf[e] = (_Float16)skip_w[(size_t)(oc0 + n) * CI_ + ks + ((e < 8) ? 0 : 16) + kb + (e & 7)];
    }
    acc = __builtin_amdgcn_wmma_f32_16x16x32_f16(false, a, false, bf,
                                                 (short)0, acc, false, false);
  }

  // add nearest-neighbor conv-map contribution (map2token), stash to LDS
  {
    int oc = oc0 + (lane & 15);
    #pragma unroll
    for (int r = 0; r < 8; ++r) {
      int mrow = r + ((lane < 16) ? 0 : 8);
      int gt = t0 + mrow;
      int b  = gt / N_;
      int n  = gt - b * N_;
      int ii = n / W_, jj = n - ii * W_;
      float cm = cmap[(((size_t)b * H2_ + (ii >> 1)) * W2_ + (jj >> 1)) * CO_ + oc];
      xbuf[mrow][oc] = acc[r] + cm;
    }
  }
  __syncthreads();

  if (tid < 16) {                      // LayerNorm stats per token
    float s0 = 0.f, s1 = 0.f;
    for (int c = 0; c < CO_; ++c) { float v = xbuf[tid][c]; s0 += v; s1 += v*v; }
    float mu  = s0 * (1.0f / CO_);
    float var = s1 * (1.0f / CO_) - mu * mu;
    mu_s[tid] = mu;
    rs_s[tid] = rsqrtf(var + LN_EPS);
  }
  __syncthreads();

  #pragma unroll
  for (int e = 0; e < 8; ++e) {        // normalize, emit f32 + bf16 copies
    int i = tid + 256 * e;
    int r = i >> 7, c = i & 127;
    float v = (xbuf[r][c] - mu_s[r]) * rs_s[r] * ln_g[c] + ln_b[c];
    xbuf[r][c] = v;
    size_t gi = (size_t)(t0 + r) * CO_ + c;
    xn_out[gi] = v;
    xnb[gi] = (__bf16)v;
  }
  __syncthreads();

  if (tid < 16) {                      // score, exp-weight, sq-norm
    float sc = 0.f, ss = 0.f;
    for (int c = 0; c < CO_; ++c) { float v = xbuf[tid][c]; sc += v * score_w[c]; ss += v*v; }
    sc += score_b[0];
    int gt = t0 + tid;
    ts_out[gt] = sc;
    weight[gt] = expf(sc);
    sq[gt] = ss;
  }
}

// ------- pairwise distances via bf16 WMMA, symmetric-tile version -------
// 64x64 tile per block; only i-tile <= j-tile blocks compute; mirrored writes
// fill the lower triangle. Each wave owns two 16x16 accumulators sharing one
// A fragment (back-to-back independent WMMAs for the matrix pipe).
__global__ void __launch_bounds__(256) CTM_dist(
    const __bf16* __restrict__ xnb, const float* __restrict__ sq,
    float* __restrict__ dist, unsigned int* __restrict__ dmax) {
  const int ti = blockIdx.y, tj = blockIdx.x;
  if (ti > tj) return;                 // uniform early-exit (below diagonal)
  __shared__ __bf16 Ash[64][CO_];      // 16 KB
  __shared__ __bf16 Bsh[64][CO_];      // 16 KB
  __shared__ float rmax[256];
  const int tid  = threadIdx.x;
  const int lane = tid & 31;
  const int wv   = tid >> 5;
  const int b    = blockIdx.z;
  const int i0b  = ti * 64;
  const int j0b  = tj * 64;

  const unsigned int* srcA = (const unsigned int*)(xnb + ((size_t)b * N_ + i0b) * CO_);
  const unsigned int* srcB = (const unsigned int*)(xnb + ((size_t)b * N_ + j0b) * CO_);
  unsigned int* dA = (unsigned int*)&Ash[0][0];
  unsigned int* dB = (unsigned int*)&Bsh[0][0];
  __builtin_prefetch(srcA, 0, 1);      // global_prefetch_b8
  __builtin_prefetch(srcB, 0, 1);
  #pragma unroll
  for (int e = 0; e < 16; ++e) dA[tid + 256 * e] = srcA[tid + 256 * e];
  #pragma unroll
  for (int e = 0; e < 16; ++e) dB[tid + 256 * e] = srcB[tid + 256 * e];
  __syncthreads();

  const int wi  = wv >> 1;             // 0..3: row 16-tile
  const int wj  = wv & 1;              // 0..1: col tiles wj*16 and wj*16+32
  const int ar0 = wi * 16;
  v8f acc0 = {}, acc1 = {};
  #pragma unroll
  for (int s = 0; s < 4; ++s) {        // K = 128 = 4 x 32
    const int ks = s * 32;
    v16bf a, b0, b1;
    {
      int m  = lane & 15;
      int kb = (lane < 16) ? 0 : 8;
      #pragma unroll
      for (int e = 0; e < 16; ++e) {
        int r = e >> 1, h = e & 1;
        int kl = (r < 4) ? (kb + 2*r + h) : (16 + kb + 2*(r-4) + h);
        a[e] = Ash[ar0 + m][ks + kl];
      }
    }
    {
      int n  = lane & 15;
      int kb = (lane < 16) ? 0 : 8;
      #pragma unroll
      for (int e = 0; e < 16; ++e) {
        int kl = ((e < 8) ? 0 : 16) + kb + (e & 7);
        b0[e] = Bsh[wj * 16 + n][ks + kl];
        b1[e] = Bsh[wj * 16 + 32 + n][ks + kl];
      }
    }
    acc0 = __builtin_amdgcn_wmma_f32_16x16x32_bf16(false, a, false, b0,
                                                   (short)0, acc0, false, false);
    acc1 = __builtin_amdgcn_wmma_f32_16x16x32_bf16(false, a, false, b1,
                                                   (short)0, acc1, false, false);
  }

  float lm = 0.f;
  const bool mir = (ti != tj);
  #pragma unroll
  for (int t = 0; t < 2; ++t) {
    int gj = j0b + wj * 16 + t * 32 + (lane & 15);
    float sqj = sq[b * N_ + gj];
    #pragma unroll
    for (int r = 0; r < 8; ++r) {
      int gi = i0b + ar0 + r + ((lane < 16) ? 0 : 8);
      float v  = t ? acc1[r] : acc0[r];
      float d2 = sq[b * N_ + gi] + sqj - 2.0f * v;
      float d  = sqrtf(fmaxf(d2, 0.f)) * INV_SQRT_C;
      dist[((size_t)b * N_ + gi) * N_ + gj] = d;
      if (mir) dist[((size_t)b * N_ + gj) * N_ + gi] = d;   // per-lane consecutive
      lm = fmaxf(lm, d);
    }
  }
  rmax[tid] = lm;
  __syncthreads();
  if (tid == 0) {
    float m0 = 0.f;
    for (int t = 0; t < 256; ++t) m0 = fmaxf(m0, rmax[t]);
    atomicMax(&dmax[b], __float_as_uint(m0));   // dist >= 0 => uint order == float order
  }
}

// ------- density: 5 smallest distances per row -------
__global__ void __launch_bounds__(64) CTM_density(
    const float* __restrict__ dist, float* __restrict__ dens) {
  __shared__ float sb[64 * K_];
  const int i = blockIdx.x;
  const float* row = dist + (size_t)i * N_;
  float best[K_] = {1e30f,1e30f,1e30f,1e30f,1e30f};
  for (int j = threadIdx.x; j < N_; j += 64) {
    float d = row[j];
    if (d < best[K_-1]) {
      best[K_-1] = d;
      #pragma unroll
      for (int q = K_-1; q > 0; --q)
        if (best[q] < best[q-1]) { float t = best[q]; best[q] = best[q-1]; best[q-1] = t; }
    }
  }
  #pragma unroll
  for (int q = 0; q < K_; ++q) sb[threadIdx.x * K_ + q] = best[q];
  __syncthreads();
  if (threadIdx.x == 0) {
    float fin[K_] = {1e30f,1e30f,1e30f,1e30f,1e30f};
    for (int t = 0; t < 64 * K_; ++t) {
      float d = sb[t];
      if (d < fin[K_-1]) {
        fin[K_-1] = d;
        #pragma unroll
        for (int q = K_-1; q > 0; --q)
          if (fin[q] < fin[q-1]) { float tt = fin[q]; fin[q] = fin[q-1]; fin[q-1] = tt; }
      }
    }
    float ms = 0.f;
    #pragma unroll
    for (int q = 0; q < K_; ++q) ms += fin[q] * fin[q];
    dens[i] = expf(-ms * (1.0f / K_));
  }
}

// ------- dist_parent: min dist to any higher-density token; score = dp*density -------
__global__ void __launch_bounds__(128) CTM_parent(
    const float* __restrict__ dist, const float* __restrict__ dens,
    const unsigned int* __restrict__ dmax, float* __restrict__ scorev) {
  __shared__ float red[128];
  const int i = blockIdx.x;
  const int b = i / N_;
  const float di = dens[i];
  const float dm = __uint_as_float(dmax[b]);
  const float* row = dist + (size_t)i * N_;
  const float* db  = dens + b * N_;
  float mn = dm;
  for (int j = threadIdx.x; j < N_; j += 128)
    if (db[j] > di) mn = fminf(mn, row[j]);
  red[threadIdx.x] = mn;
  __syncthreads();
  for (int s = 64; s > 0; s >>= 1) {
    if (threadIdx.x < s) red[threadIdx.x] = fminf(red[threadIdx.x], red[threadIdx.x + s]);
    __syncthreads();
  }
  if (threadIdx.x == 0) scorev[i] = red[0] * di;
}

// ------- exact top-M by rank counting (tie-break: lower index first, like top_k) -------
__global__ void CTM_rank(const float* __restrict__ scorev, int* __restrict__ idown) {
  int i = blockIdx.x * blockDim.x + threadIdx.x;
  if (i >= B_ * N_) return;
  int b = i / N_, n = i - b * N_;
  const float* sb = scorev + b * N_;
  float s = sb[n];
  int cnt = 0;
  for (int j = 0; j < N_; ++j) {
    float sj = sb[j];
    cnt += (sj > s) || (sj == s && j < n);
  }
  if (cnt < M_) idown[b * M_ + cnt] = n;
}

// ------- assign each token to nearest center (first-wins argmin) -------
__global__ void CTM_assign(const float* __restrict__ dist, const int* __restrict__ idown,
                           int* __restrict__ idx_out) {
  int i = blockIdx.x * blockDim.x + threadIdx.x;
  if (i >= B_ * N_) return;
  int b = i / N_, n = i - b * N_;
  float best = 1e30f; int bm = 0;
  for (int m = 0; m < M_; ++m) {
    int c = idown[b * M_ + m];
    float d = dist[((size_t)b * N_ + c) * N_ + n];
    if (d < best) { best = d; bm = m; }
  }
  idx_out[i] = bm;
}

__global__ void CTM_centers(const int* __restrict__ idown, int* __restrict__ idx_out) {
  int t = blockIdx.x * blockDim.x + threadIdx.x;
  if (t >= B_ * M_) return;
  int b = t / M_, m = t - b * M_;
  idx_out[b * N_ + idown[t]] = m;
}

// ------- weighted segment sums -------
__global__ void CTM_allw(const int* __restrict__ idx, const float* __restrict__ w,
                         float* __restrict__ allw) {
  int i = blockIdx.x * blockDim.x + threadIdx.x;
  if (i >= B_ * N_) return;
  int b = i / N_;
  atomicAdd(&allw[b * M_ + idx[i]], w[i]);
}

__global__ void CTM_merge(const int* __restrict__ idx, const float* __restrict__ w,
                          const float* __restrict__ allw, const float* __restrict__ xn,
                          float* __restrict__ xm) {
  int t = blockIdx.x * blockDim.x + threadIdx.x;
  if (t >= B_ * N_ * CO_) return;
  int i = t >> 7, c = t & 127;
  int b = i / N_;
  int seg = idx[i];
  float nw = w[i] / (allw[b * M_ + seg] + 1e-6f);
  atomicAdd(&xm[((size_t)b * M_ + seg) * CO_ + c], xn[t] * nw);
}

extern "C" void kernel_launch(void* const* d_in, const int* in_sizes, int n_in,
                              void* d_out, int out_size, void* d_ws, size_t ws_size,
                              hipStream_t stream) {
  (void)in_sizes; (void)n_in; (void)out_size; (void)ws_size;
  const float* x       = (const float*)d_in[0];
  const float* skip_w  = (const float*)d_in[1];
  const float* conv_w  = (const float*)d_in[2];
  const float* conv_b  = (const float*)d_in[3];
  const float* ln_g    = (const float*)d_in[4];
  const float* ln_b    = (const float*)d_in[5];
  const float* score_w = (const float*)d_in[6];
  const float* score_b = (const float*)d_in[7];

  float* out  = (float*)d_out;
  float* xm   = out + OUT_XM;
  float* xn   = out + OUT_XN;
  float* ts   = out + OUT_TS;
  int*   idxc = (int*)(out + OUT_IDX);

  float* wsf = (float*)d_ws;
  float*        cmap   = wsf + OFF_CONV;
  __bf16*       xnb    = (__bf16*)(wsf + OFF_XNB);
  float*        sq     = wsf + OFF_SQ;
  float*        weight = wsf + OFF_WEIGHT;
  float*        dens   = wsf + OFF_DENS;
  unsigned int* dmax   = (unsigned int*)(wsf + OFF_DMAX);
  float*        scorev = wsf + OFF_SCORE;
  int*          idown  = (int*)(wsf + OFF_IDOWN);
  float*        allw   = wsf + OFF_ALLW;
  float*        dist   = wsf + OFF_DIST;

  CTM_init<<<(B_*M_*CO_ + 255) / 256, 256, 0, stream>>>(xm, allw, dmax);
  CTM_conv<<<(B_*H2_*W2_*CO_ + 255) / 256, 256, 0, stream>>>(x, conv_w, conv_b, cmap);
  CTM_fused_ln<<<(B_*N_) / 16, 256, 0, stream>>>(x, skip_w, cmap, ln_g, ln_b,
                                                 score_w, score_b, xn, ts, xnb, sq, weight);
  dim3 g2(N_ / 64, N_ / 64, B_);       // 49 x 49 x 4; sub-diagonal blocks exit
  CTM_dist<<<g2, 256, 0, stream>>>(xnb, sq, dist, dmax);
  CTM_density<<<B_*N_, 64, 0, stream>>>(dist, dens);
  CTM_parent<<<B_*N_, 128, 0, stream>>>(dist, dens, dmax, scorev);
  CTM_rank<<<(B_*N_ + 255) / 256, 256, 0, stream>>>(scorev, idown);
  CTM_assign<<<(B_*N_ + 255) / 256, 256, 0, stream>>>(dist, idown, idxc);
  CTM_centers<<<(B_*M_ + 255) / 256, 256, 0, stream>>>(idown, idxc);
  CTM_allw<<<(B_*N_ + 255) / 256, 256, 0, stream>>>(idxc, weight, allw);
  CTM_merge<<<(B_*N_*CO_ + 255) / 256, 256, 0, stream>>>(idxc, weight, allw, xn, xm);
}
